// S4_45543833207427
// MI455X (gfx1250) — compile-verified, matrix operand order checked by reading, not run on hardware
//
#include <hip/hip_runtime.h>
#include <hip/hip_fp16.h>

// Problem constants (from reference): B=32, L=4096, D_IN=8, H=512, N=32, D_OUT=1
#define BB    32
#define LL    4096
#define DIN   8
#define HH    512
#define HC    256     // H/2  (also first-layer width)
#define NN    32
#define VSTR  264     // LDS row stride (halves) for v tile: 256 + 8 pad -> conflict-free

typedef __attribute__((ext_vector_type(16))) _Float16 v16h;
typedef __attribute__((ext_vector_type(8)))  float    v8f;

union FragH { v16h h; unsigned int u[8]; };

// ---------------------------------------------------------------------------
// 1) Cbar[h,n] = (C_re + i C_im) * (exp(dt*A) - 1) / A       (16K elements)
// ---------------------------------------------------------------------------
__global__ void k_cbar(const float* __restrict__ log_dt,
                       const float* __restrict__ A_real,
                       const float* __restrict__ A_imag,
                       const float* __restrict__ C_re,
                       const float* __restrict__ C_im,
                       float* __restrict__ Cbr, float* __restrict__ Cbi) {
    int idx = blockIdx.x * blockDim.x + threadIdx.x;   // h*32 + n
    if (idx >= HH * NN) return;
    int h = idx >> 5;
    float dt = __expf(log_dt[h]);
    float ar = A_real[idx], ai = A_imag[idx];
    float er = __expf(dt * ar);
    float s, c; __sincosf(dt * ai, &s, &c);
    float e1r = er * c - 1.0f;                         // exp(dtA) - 1
    float e1i = er * s;
    float den = 1.0f / (ar * ar + ai * ai);            // divide by A = *conj(A)/|A|^2
    float qr = (e1r * ar + e1i * ai) * den;
    float qi = (e1i * ar - e1r * ai) * den;
    float cr = C_re[idx], ci = C_im[idx];
    Cbr[idx] = cr * qr - ci * qi;
    Cbi[idx] = cr * qi + ci * qr;
}

// ---------------------------------------------------------------------------
// 2) W_in2 (256x512 f32, [c][h]) -> Wt (f16, transposed [h][c]) so that the
//    WMMA B-fragment's adjacent K pair is a single 32-bit load.
// ---------------------------------------------------------------------------
__global__ void k_wt(const float* __restrict__ W_in2, __half* __restrict__ Wt) {
    int idx = blockIdx.x * blockDim.x + threadIdx.x;   // h*256 + c  (coalesced writes)
    if (idx >= HC * HH) return;
    int h = idx >> 8, c = idx & 255;
    Wt[idx] = __float2half(W_in2[c * HH + h]);
}

__global__ void k_zero(float* __restrict__ acc) {
    int idx = blockIdx.x * blockDim.x + threadIdx.x;
    if (idx < BB * HH) acc[idx] = 0.0f;
}

// ---------------------------------------------------------------------------
// 3) Kr[t][h] = K[h, L-1-t] = 2 * Re( sum_n Cbar[h,n] * exp(dtA[h,n]*(L-1-t)) )
//    One block per t (512 threads = h). Stores perfectly coalesced; Cbar/A are
//    64 KB arrays -> L2-hot for the strided per-lane reads.
// ---------------------------------------------------------------------------
__global__ void k_krev(const float* __restrict__ log_dt,
                       const float* __restrict__ A_real,
                       const float* __restrict__ A_imag,
                       const float* __restrict__ Cbr,
                       const float* __restrict__ Cbi,
                       float* __restrict__ Kr) {
    int t = blockIdx.x;            // 0..4095
    int h = threadIdx.x;           // 0..511
    float l  = (float)(LL - 1 - t);
    float dtl = __expf(log_dt[h]) * l;
    const float* ar = A_real + h * NN;
    const float* ai = A_imag + h * NN;
    const float* br = Cbr    + h * NN;
    const float* bi = Cbi    + h * NN;
    float sum = 0.0f;
#pragma unroll 8
    for (int n = 0; n < NN; ++n) {
        float er = __expf(ar[n] * dtl);
        float s, c; __sincosf(ai[n] * dtl, &s, &c);
        sum = fmaf(br[n],  er * c, sum);
        sum = fmaf(-bi[n], er * s, sum);
    }
    Kr[t * HH + h] = 2.0f * sum;
}

// ---------------------------------------------------------------------------
// 4) Fused: u = relu(X@W_in1 + b1) @ W_in2 + b2  (WMMA f16->f32), then
//    acc[b,h] += sum_t u[b,t,h] * Kr[t,h];  also captures u at t = L-1.
//    Grid: (L/32, B) x 256 threads (8 waves). Workgroup tile: 32 t-rows x 512 h.
//    Each wave owns a 64-wide h slice (4 tiles) and BOTH 16-row t-tiles, so
//    every B fragment feeds 2 WMMAs; all 8 B fragments of a K-sweep are
//    preloaded so the loads clause and WMMAs pipeline against loadcnt.
// ---------------------------------------------------------------------------
__global__ void __launch_bounds__(256)
k_main(const float* __restrict__ X,
       const float* __restrict__ W_in1, const float* __restrict__ b_in1,
       const __half* __restrict__ Wt,   const float* __restrict__ b_in2,
       const float* __restrict__ Kr,
       float* __restrict__ acc, float* __restrict__ uLast) {
    __shared__ float  Xs[32 * DIN];           // 32 rows of X
    __shared__ __half vLds[32 * VSTR];        // relu(X@W_in1+b1), f16, padded rows

    const int tid   = threadIdx.x;
    const int bIdx  = blockIdx.y;             // batch
    const int tBase = blockIdx.x * 32;        // first t row of this workgroup

    // ---- stage 0: load X tile (32x8 f32, contiguous) --------------------
    if (tid < 32 * DIN)
        Xs[tid] = X[(bIdx * LL + tBase) * DIN + tid];
    __syncthreads();

    // ---- stage 1: v[t][c] = relu(X@W_in1 + b1), store f16 to LDS --------
    {
        const int c = tid;                    // 0..255
        float wcol[DIN];
#pragma unroll
        for (int k = 0; k < DIN; ++k) wcol[k] = W_in1[k * HC + c];
        const float bias1 = b_in1[c];
#pragma unroll 4
        for (int t = 0; t < 32; ++t) {
            float a = bias1;
#pragma unroll
            for (int k = 0; k < DIN; ++k) a = fmaf(Xs[t * DIN + k], wcol[k], a);
            vLds[t * VSTR + c] = __float2half(fmaxf(a, 0.0f));
        }
    }
    __syncthreads();

    // ---- stage 2: WMMA matmul + conv reduction --------------------------
    const int wave    = tid >> 5;
    const int lane    = tid & 31;
    const int Mn      = lane & 15;            // A row / B,C col within tile
    const int halfSel = lane >> 4;            // lane half (ISA layout)

    for (int hTile = 0; hTile < 4; ++hTile) {
        const int hbase = wave * 64 + hTile * 16;

        // Preload all 8 B fragments of the K=256 sweep (16x b128 loads clause)
        FragH bfr[8];
#pragma unroll
        for (int cs = 0; cs < 8; ++cs) {
#pragma unroll
            for (int j = 0; j < 8; ++j) {
                // 16-bit B 32x16 layout: VGPR j holds K pair {k0,k0+1},
                // k0 = 2j (+16 for j>=4) + 8*laneHalf  (ISA 7.12.2 mirrored)
                const int k0 = ((j < 4) ? (2 * j) : (16 + 2 * (j - 4))) + halfSel * 8;
                bfr[cs].u[j] = *(const unsigned int*)&Wt[(hbase + Mn) * HC + cs * 32 + k0];
            }
        }

        v8f c0 = {};                          // rows tBase .. tBase+15
        v8f c1 = {};                          // rows tBase+16 .. tBase+31
#pragma unroll
        for (int cs = 0; cs < 8; ++cs) {
            FragH a0, a1;
#pragma unroll
            for (int j = 0; j < 8; ++j) {
                const int k0 = ((j < 4) ? (2 * j) : (16 + 2 * (j - 4))) + halfSel * 8;
                a0.u[j] = *(const unsigned int*)&vLds[Mn        * VSTR + cs * 32 + k0];
                a1.u[j] = *(const unsigned int*)&vLds[(16 + Mn) * VSTR + cs * 32 + k0];
            }
            c0 = __builtin_amdgcn_wmma_f32_16x16x32_f16(
                false, a0.h, false, bfr[cs].h, (short)0, c0, false, false);
            c1 = __builtin_amdgcn_wmma_f32_16x16x32_f16(
                false, a1.h, false, bfr[cs].h, (short)0, c1, false, false);
        }

        // bias + weighted row-reduction against Kr, then cross-half combine
        const float bias2 = b_in2[hbase + Mn];
        float part = 0.0f;
#pragma unroll
        for (int r = 0; r < 8; ++r) {         // C/D layout: M = r + 8*laneHalf
            const float uv0 = c0[r] + bias2;
            const float uv1 = c1[r] + bias2;
            const int  trow0 = tBase + r + halfSel * 8;       // <= 4079, never L-1
            const int  trow1 = trow0 + 16;
            part = fmaf(uv0, Kr[trow0 * HH + hbase + Mn], part);
            part = fmaf(uv1, Kr[trow1 * HH + hbase + Mn], part);
            if (trow1 == LL - 1) uLast[bIdx * HH + hbase + Mn] = uv1;
        }
        part += __shfl_xor(part, 16, 32);     // wave32: combine the two halves
        if (halfSel == 0)
            atomicAdd(&acc[bIdx * HH + hbase + Mn], part);
    }
}

// ---------------------------------------------------------------------------
// 5) Last-step tail: gelu(conv + u*D) -> GLU -> relu MLP -> sigmoid. 1 block/b.
// ---------------------------------------------------------------------------
__global__ void __launch_bounds__(512)
k_finish(const float* __restrict__ acc, const float* __restrict__ uLast,
         const float* __restrict__ D,
         const float* __restrict__ W_glu,  const float* __restrict__ b_glu,
         const float* __restrict__ W_out1, const float* __restrict__ b_out1,
         const float* __restrict__ W_out2, const float* __restrict__ b_out2,
         float* __restrict__ out) {
    __shared__ float sy[HH];
    __shared__ float gl[HH];
    __shared__ float red[HC];
    const int b = blockIdx.x, tid = threadIdx.x;

    // y = gelu(conv + u_last * D)   (tanh approximation, jax default)
    {
        float y = acc[b * HH + tid] + uLast[b * HH + tid] * D[tid];
        float t = tanhf(0.7978845608028654f * (y + 0.044715f * y * y * y));
        sy[tid] = 0.5f * y * (1.0f + t);
    }
    __syncthreads();

    // GLU: z = y@W_glu + b_glu;  g = z[:512] * sigmoid(z[512:])
    {
        float z1 = b_glu[tid], z2 = b_glu[tid + HH];
        for (int hp = 0; hp < HH; ++hp) {
            const float yv = sy[hp];
            z1 = fmaf(yv, W_glu[hp * (2 * HH) + tid],      z1);
            z2 = fmaf(yv, W_glu[hp * (2 * HH) + tid + HH], z2);
        }
        gl[tid] = z1 * (1.0f / (1.0f + __expf(-z2)));
    }
    __syncthreads();

    // r1 = relu(g @ W_out1 + b_out1)   (256 outputs), folded with W_out2 (256x1)
    if (tid < HC) {
        float r = b_out1[tid];
        for (int hp = 0; hp < HH; ++hp)
            r = fmaf(gl[hp], W_out1[hp * HC + tid], r);
        red[tid] = fmaxf(r, 0.0f) * W_out2[tid];
    }
    __syncthreads();
    for (int s = HC / 2; s > 0; s >>= 1) {
        if (tid < s) red[tid] += red[tid + s];
        __syncthreads();
    }
    if (tid == 0) {
        const float o = red[0] + b_out2[0];
        out[b] = 1.0f / (1.0f + __expf(-o));
    }
}

// ---------------------------------------------------------------------------
extern "C" void kernel_launch(void* const* d_in, const int* in_sizes, int n_in,
                              void* d_out, int out_size, void* d_ws, size_t ws_size,
                              hipStream_t stream) {
    const float* X      = (const float*)d_in[0];
    const float* W_in1  = (const float*)d_in[1];
    const float* b_in1  = (const float*)d_in[2];
    const float* W_in2  = (const float*)d_in[3];
    const float* b_in2  = (const float*)d_in[4];
    const float* log_dt = (const float*)d_in[5];
    const float* A_real = (const float*)d_in[6];
    const float* A_imag = (const float*)d_in[7];
    const float* C_re   = (const float*)d_in[8];
    const float* C_im   = (const float*)d_in[9];
    const float* Dvec   = (const float*)d_in[10];
    const float* W_glu  = (const float*)d_in[11];
    const float* b_glu  = (const float*)d_in[12];
    const float* W_out1 = (const float*)d_in[13];
    const float* b_out1 = (const float*)d_in[14];
    const float* W_out2 = (const float*)d_in[15];
    const float* b_out2 = (const float*)d_in[16];
    float* out = (float*)d_out;

    // workspace layout (~8.9 MB)
    char* w = (char*)d_ws;
    float*  Kr    = (float*)w;                         w += (size_t)LL * HH * 4;  // 8 MB
    float*  Cbr   = (float*)w;                         w += HH * NN * 4;
    float*  Cbi   = (float*)w;                         w += HH * NN * 4;
    __half* Wt    = (__half*)w;                        w += HC * HH * 2;
    float*  acc   = (float*)w;                         w += BB * HH * 4;
    float*  uLast = (float*)w;                         w += BB * HH * 4;
    (void)ws_size; (void)in_sizes; (void)n_in; (void)out_size;

    k_cbar<<<dim3((HH * NN + 255) / 256), dim3(256), 0, stream>>>(
        log_dt, A_real, A_imag, C_re, C_im, Cbr, Cbi);
    k_wt<<<dim3((HC * HH + 255) / 256), dim3(256), 0, stream>>>(W_in2, Wt);
    k_zero<<<dim3((BB * HH + 511) / 512), dim3(512), 0, stream>>>(acc);
    k_krev<<<dim3(LL), dim3(HH), 0, stream>>>(log_dt, A_real, A_imag, Cbr, Cbi, Kr);
    k_main<<<dim3(LL / 32, BB), dim3(256), 0, stream>>>(
        X, W_in1, b_in1, Wt, b_in2, Kr, acc, uLast);
    k_finish<<<dim3(BB), dim3(512), 0, stream>>>(
        acc, uLast, Dvec, W_glu, b_glu, W_out1, b_out1, W_out2, b_out2, out);
}
